// EmoAttention_7851200217161
// MI455X (gfx1250) — compile-verified, matrix operand order checked by reading, not run on hardware
//
#include <hip/hip_runtime.h>
#include <hip/hip_bf16.h>
#include <stdint.h>

typedef __attribute__((ext_vector_type(16))) __bf16 v16bf;
typedef __attribute__((ext_vector_type(8)))  __bf16 bf16x8;
typedef __attribute__((ext_vector_type(4)))  __bf16 bf16x4;
typedef __attribute__((ext_vector_type(8)))  float  v8f;
typedef __attribute__((address_space(3)))    __bf16 lds_bf16;

#define BLK_M 128
#define BLK_N 128
#define BLK_K 32
#define LDPAD 40   // LDS row stride in bf16: 80 bytes -> 16B-aligned rows for b128 ds ops

// Load a 16x32 bf16 fragment (A-style layout, CDNA5 ISA 7.12.2) from an LDS tile
// stored as tile[row][k], row stride LDPAD.
//   lane 0-15  : row = row_base+lane,    k = {0..7, 16..23}
//   lane 16-31 : row = row_base+lane-16, k = {8..15, 24..31}
__device__ __forceinline__ v16bf load_frag_lds(const __bf16* tile, int row_base) {
  const int lane = threadIdx.x & 31;
  const int half = lane >> 4;
  const int r    = lane & 15;
  const __bf16* p = tile + (row_base + r) * LDPAD + half * 8;
  v16bf f;
#pragma unroll
  for (int i = 0; i < 8; ++i) f[i] = p[i];
#pragma unroll
  for (int i = 0; i < 8; ++i) f[8 + i] = p[16 + i];
  return f;
}

// Stage 16 consecutive f32 -> 16 bf16 into LDS (two 16B ds stores).
__device__ __forceinline__ void stage_f32(__bf16* dst, const float* src) {
  float4 f0 = *(const float4*)(src);
  float4 f1 = *(const float4*)(src + 4);
  float4 f2 = *(const float4*)(src + 8);
  float4 f3 = *(const float4*)(src + 12);
  bf16x8 lo, hi;
  lo[0] = (__bf16)f0.x; lo[1] = (__bf16)f0.y; lo[2] = (__bf16)f0.z; lo[3] = (__bf16)f0.w;
  lo[4] = (__bf16)f1.x; lo[5] = (__bf16)f1.y; lo[6] = (__bf16)f1.z; lo[7] = (__bf16)f1.w;
  hi[0] = (__bf16)f2.x; hi[1] = (__bf16)f2.y; hi[2] = (__bf16)f2.z; hi[3] = (__bf16)f2.w;
  hi[4] = (__bf16)f3.x; hi[5] = (__bf16)f3.y; hi[6] = (__bf16)f3.z; hi[7] = (__bf16)f3.w;
  *(bf16x8*)(dst)     = lo;
  *(bf16x8*)(dst + 8) = hi;
}

// Async global->LDS copy of 32 bytes (2 x b128), no VGPR round-trip.
// IOFFSET is added to BOTH the LDS and global addresses (ISA 10.4), so a single
// per-lane (ldsaddr, gaddr) pair covers both chunks. Tracked by ASYNCcnt.
__device__ __forceinline__ void async_stage32B(lds_bf16* dst, const __bf16* src) {
  asm volatile(
      "global_load_async_to_lds_b128 %0, %1, off\n"
      "global_load_async_to_lds_b128 %0, %1, off offset:16"
      :: "v"(dst), "v"(src) : "memory");
}

__device__ __forceinline__ void wait_async0() {
  asm volatile("s_wait_asynccnt 0x0" ::: "memory");
}

// Uniform WMMA GEMM: C[bz] = alpha * A[bz] x B[bz]^T
//  A: [M,K] row-major, f32 (converted while staging) or bf16 (async-copied)
//  B: [N,K] row-major bf16 (async-copied into LDS)
//  C: bf16 or f32 (f32 optionally duplicated to dup1/dup2); OUT_TRANS stores C^T.
// 256 threads = 8 waves arranged 4(m) x 2(n); wave tile 32x64 = 2x4 WMMA tiles.
// LDS double-buffered; next tile staged (async where possible) under the WMMAs.
template <bool A_F32, bool OUT_BF16, bool OUT_TRANS>
__global__ __launch_bounds__(256)
void wmma_gemm(const void* __restrict__ Ap, const __bf16* __restrict__ Bp,
               void* __restrict__ Cp, float* __restrict__ dup1, float* __restrict__ dup2,
               int K, int lda, int ldb, int ldc,
               long long bsA, long long bsB, long long bsC, float alpha) {
  __shared__ __align__(16) __bf16 As[2][BLK_M * LDPAD];
  __shared__ __align__(16) __bf16 Bs[2][BLK_N * LDPAD];

  const int t    = threadIdx.x;
  const int wave = t >> 5;
  const int wm   = wave >> 1;   // 0..3 -> 32-row sub-tile
  const int wn   = wave & 1;    // 0..1 -> 64-col sub-tile
  const int bz   = blockIdx.z;
  const int m0   = blockIdx.y * BLK_M;
  const int n0   = blockIdx.x * BLK_N;

  // staging: thread t owns 16 consecutive elements of one tile row
  const int srow = t >> 1;
  const int scg  = (t & 1) * 16;
  const int soff = srow * LDPAD + scg;

  const float*  aptrf = (const float*)Ap  + (size_t)bz * (size_t)bsA + (size_t)(m0 + srow) * lda + scg;
  const __bf16* aptrb = (const __bf16*)Ap + (size_t)bz * (size_t)bsA + (size_t)(m0 + srow) * lda + scg;
  const __bf16* bptr  = Bp                + (size_t)bz * (size_t)bsB + (size_t)(n0 + srow) * ldb + scg;

  v8f acc[8];
#pragma unroll
  for (int i = 0; i < 8; ++i) {
    v8f z = {0.f, 0.f, 0.f, 0.f, 0.f, 0.f, 0.f, 0.f};
    acc[i] = z;
  }

  // prologue: stage k-tile 0 into buffer 0
  if (A_F32) { stage_f32(As[0] + soff, aptrf); aptrf += BLK_K; }
  else       { async_stage32B((lds_bf16*)(As[0] + soff), aptrb); aptrb += BLK_K; }
  async_stage32B((lds_bf16*)(Bs[0] + soff), bptr); bptr += BLK_K;
  wait_async0();
  __syncthreads();

  int buf = 0;
  for (int kb = 0; kb < K; kb += BLK_K) {
    // stage next k-tile into the other buffer (runs under the WMMAs below)
    if (kb + BLK_K < K) {
      if (A_F32) {
        stage_f32(As[buf ^ 1] + soff, aptrf); aptrf += BLK_K;
        __builtin_prefetch(aptrf, 0, 1);  // global_prefetch_b8, tile k+2
      } else {
        async_stage32B((lds_bf16*)(As[buf ^ 1] + soff), aptrb); aptrb += BLK_K;
      }
      async_stage32B((lds_bf16*)(Bs[buf ^ 1] + soff), bptr); bptr += BLK_K;
    }

    // compute current buffer: 2 A frags + 4 B frags -> 8 WMMAs
    v16bf a0 = load_frag_lds(As[buf], wm * 32);
    v16bf a1 = load_frag_lds(As[buf], wm * 32 + 16);
#pragma unroll
    for (int nt = 0; nt < 4; ++nt) {
      v16bf b = load_frag_lds(Bs[buf], wn * 64 + nt * 16);
      acc[nt]     = __builtin_amdgcn_wmma_f32_16x16x32_bf16(
          false, a0, false, b, (short)0, acc[nt], false, false);
      acc[4 + nt] = __builtin_amdgcn_wmma_f32_16x16x32_bf16(
          false, a1, false, b, (short)0, acc[4 + nt], false, false);
    }
    wait_async0();     // next tile's async LDS writes have landed
    __syncthreads();   // single barrier per k-step (double buffered)
    buf ^= 1;
  }

  // ---- epilogue: C/D layout (VGPR j: lanes0-15 M=j, lanes16-31 M=j+8; N=lane%16)
  const int lane = t & 31, half = lane >> 4, r = lane & 15;
#pragma unroll
  for (int a = 0; a < 2; ++a) {
#pragma unroll
    for (int nt = 0; nt < 4; ++nt) {
#pragma unroll
      for (int j = 0; j < 8; ++j) {
        int row = m0 + wm * 32 + a * 16 + half * 8 + j;
        int col = n0 + wn * 64 + nt * 16 + r;
        float v = alpha * acc[a * 4 + nt][j];
        if (OUT_TRANS) {
          ((__bf16*)Cp)[(size_t)bz * (size_t)bsC + (size_t)col * ldc + row] = (__bf16)v;
        } else if (OUT_BF16) {
          ((__bf16*)Cp)[(size_t)bz * (size_t)bsC + (size_t)row * ldc + col] = (__bf16)v;
        } else {
          size_t idx = (size_t)bz * (size_t)bsC + (size_t)row * ldc + col;
          ((float*)Cp)[idx] = v;
          if (dup1) dup1[idx] = v;
          if (dup2) dup2[idx] = v;
        }
      }
    }
  }
}

// LDS-tiled transpose + f32->bf16 convert: dst[c][r] = scale * src[r][c]
// src: [R,C] row-major f32 ; dst: [C,R] row-major bf16 ; 32x32 tiles, 256 threads.
__global__ __launch_bounds__(256)
void trans_cvt(const float* __restrict__ src, __bf16* __restrict__ dst,
               int R, int C, long long bsSrc, long long bsDst, float scale) {
  __shared__ float tile[32][33];
  const int bz = blockIdx.z;
  const float* s = src + (size_t)bz * (size_t)bsSrc;
  __bf16* d = dst + (size_t)bz * (size_t)bsDst;
  const int r0 = blockIdx.y * 32, c0 = blockIdx.x * 32;
  const int tx = threadIdx.x & 31, ty = threadIdx.x >> 5;
#pragma unroll
  for (int i = 0; i < 4; ++i)
    tile[ty + i * 8][tx] = s[(size_t)(r0 + ty + i * 8) * C + c0 + tx];
  __syncthreads();
#pragma unroll
  for (int i = 0; i < 4; ++i)
    d[(size_t)(c0 + ty + i * 8) * R + r0 + tx] = (__bf16)(tile[tx][ty + i * 8] * scale);
}

// emo halves of concatenated Q/K (bf16, Q scaled) + f32 emo pass-through output.
__global__ __launch_bounds__(256)
void fill_emo(const float* __restrict__ emo,
              __bf16* __restrict__ Qb, __bf16* __restrict__ Kb,
              float* __restrict__ out_emo, int total4, float scale) {
  int i = blockIdx.x * 256 + threadIdx.x;
  if (i >= total4) return;
  int i4 = i * 4;
  int d = i4 & 1023;              // DK = 1024
  size_t rs = (size_t)(i4 >> 10); // row index b*S+s
  float4 e = *(const float4*)(emo + i4);
  size_t o = rs * 2048 + 1024 + d;
  bf16x4 q, k;
  q[0] = (__bf16)(e.x * scale); q[1] = (__bf16)(e.y * scale);
  q[2] = (__bf16)(e.z * scale); q[3] = (__bf16)(e.w * scale);
  k[0] = (__bf16)e.x; k[1] = (__bf16)e.y; k[2] = (__bf16)e.z; k[3] = (__bf16)e.w;
  *(bf16x4*)(Qb + o) = q;
  *(bf16x4*)(Kb + o) = k;
  *(float4*)(out_emo + i4) = e;
}

__global__ __launch_bounds__(256)
void mask_copy(const int* __restrict__ mask, float* __restrict__ outm, int n) {
  int i = blockIdx.x * 256 + threadIdx.x;
  if (i < n) outm[i] = mask[i] ? 1.0f : 0.0f;
}

// masked row softmax over S=2048 cols, 256 threads/row, writes bf16 probs
__global__ __launch_bounds__(256)
void softmax_rows(const float* __restrict__ Lg, const int* __restrict__ mask,
                  __bf16* __restrict__ P, int S) {
  const int b = blockIdx.y, row = blockIdx.x, t = threadIdx.x;
  const float* L = Lg + ((size_t)b * S + row) * (size_t)S;
  const int* mrow = mask + (size_t)b * S;
  __bf16* Pr = P + ((size_t)b * S + row) * (size_t)S;

  float vals[8];
  float mx = -3.0e38f;
#pragma unroll
  for (int i = 0; i < 8; ++i) {
    int c = t + i * 256;
    float v = mrow[c] ? -1.0e18f : L[c];
    vals[i] = v;
    mx = fmaxf(mx, v);
  }
  __shared__ float red[8];
#pragma unroll
  for (int m = 16; m > 0; m >>= 1) mx = fmaxf(mx, __shfl_xor(mx, m, 32));
  if ((t & 31) == 0) red[t >> 5] = mx;
  __syncthreads();
  float bm = red[0];
#pragma unroll
  for (int i = 1; i < 8; ++i) bm = fmaxf(bm, red[i]);

  float s = 0.f;
#pragma unroll
  for (int i = 0; i < 8; ++i) {
    float e = __expf(vals[i] - bm);
    vals[i] = e;
    s += e;
  }
#pragma unroll
  for (int m = 16; m > 0; m >>= 1) s += __shfl_xor(s, m, 32);
  __syncthreads();
  if ((t & 31) == 0) red[t >> 5] = s;
  __syncthreads();
  float tot = 0.f;
#pragma unroll
  for (int i = 0; i < 8; ++i) tot += red[i];
  float inv = 1.0f / tot;
#pragma unroll
  for (int i = 0; i < 8; ++i) {
    int c = t + i * 256;
    Pr[c] = (__bf16)(vals[i] * inv);
  }
}

extern "C" void kernel_launch(void* const* d_in, const int* in_sizes, int n_in,
                              void* d_out, int out_size, void* d_ws, size_t ws_size,
                              hipStream_t stream) {
  (void)in_sizes; (void)n_in; (void)out_size; (void)ws_size;
  constexpr int B = 4, S = 2048, DIN = 1024, DK = 1024, DM = 2048, DOUT = 1024;
  const float scale = 0.03125f;  // DK^-0.5 = 1/32

  const float* queries = (const float*)d_in[0];
  const float* emo     = (const float*)d_in[1];
  const float* keys    = (const float*)d_in[2];
  const float* values  = (const float*)d_in[3];
  const int*   mask    = (const int*)  d_in[4];
  const float* Wq      = (const float*)d_in[5];
  const float* Wk      = (const float*)d_in[6];
  const float* Wv      = (const float*)d_in[7];
  const float* Wo      = (const float*)d_in[8];

  // workspace: Qb/Kb [B,S,DM] bf16, Vt [B,DM,S] bf16, Ctx [B,S,DM] bf16,
  //            Pb [B,S,S] bf16, Lg [B,S,S] f32, transposed bf16 weights. ~245MB.
  const size_t nBSDM = (size_t)B * S * DM;  // 16,777,216
  __bf16* Qb  = (__bf16*)d_ws;
  __bf16* Kb  = Qb  + nBSDM;
  __bf16* Vt  = Kb  + nBSDM;        // [B][DM][S] (transposed V)
  __bf16* Ctx = Vt  + nBSDM;
  __bf16* Pb  = Ctx + nBSDM;
  float*  Lg  = (float*)(Pb + nBSDM);
  __bf16* Wqt = (__bf16*)(Lg + (size_t)B * S * S);  // [DK][DIN]
  __bf16* Wkt = Wqt + (size_t)DK * DIN;
  __bf16* Wvt = Wkt + (size_t)DK * DIN;
  __bf16* Wot = Wvt + (size_t)DK * DIN;             // [DOUT][DM]

  float* out0 = (float*)d_out;                      // outputs
  float* out1 = out0 + (size_t)B * S * DOUT;        // emo copy
  float* out2 = out1 + (size_t)B * S * DK;          // outputs (dup)
  float* out3 = out2 + (size_t)B * S * DOUT;        // outputs (dup)
  float* outm = out3 + (size_t)B * S * DOUT;        // mask as float

  dim3 blk(256);

  // 0) transpose+convert weights to bf16 [N,K]; emo -> upper half of Vt
  trans_cvt<<<dim3(DK / 32, DIN / 32, 1), blk, 0, stream>>>(Wq, Wqt, DIN, DK, 0, 0, 1.0f);
  trans_cvt<<<dim3(DK / 32, DIN / 32, 1), blk, 0, stream>>>(Wk, Wkt, DIN, DK, 0, 0, 1.0f);
  trans_cvt<<<dim3(DK / 32, DIN / 32, 1), blk, 0, stream>>>(Wv, Wvt, DIN, DK, 0, 0, 1.0f);
  trans_cvt<<<dim3(DOUT / 32, DM / 32, 1), blk, 0, stream>>>(Wo, Wot, DM, DOUT, 0, 0, 1.0f);
  trans_cvt<<<dim3(DK / 32, S / 32, B), blk, 0, stream>>>(
      emo, Vt + (size_t)DK * S, S, DK,
      (long long)S * DK, (long long)DM * S, 1.0f);

  // 1) projections (A f32, B = Wt bf16 [N,K])
  wmma_gemm<true, true, false><<<dim3(DK / 128, (B * S) / 128, 1), blk, 0, stream>>>(
      queries, Wqt, Qb, nullptr, nullptr, DIN, DIN, DIN, DM, 0, 0, 0, scale);
  wmma_gemm<true, true, false><<<dim3(DK / 128, (B * S) / 128, 1), blk, 0, stream>>>(
      keys, Wkt, Kb, nullptr, nullptr, DIN, DIN, DIN, DM, 0, 0, 0, 1.0f);
  // V projection written transposed into lower half of Vt (per batch)
  wmma_gemm<true, true, true><<<dim3(DK / 128, S / 128, B), blk, 0, stream>>>(
      values, Wvt, Vt, nullptr, nullptr, DIN, DIN, DIN, S,
      (long long)S * DIN, 0, (long long)DM * S, 1.0f);

  // 2) emo halves of Q/K + pass-through outputs
  fill_emo<<<dim3((B * S * DK / 4 + 255) / 256), blk, 0, stream>>>(
      emo, Qb, Kb, out1, B * S * DK / 4, scale);
  mask_copy<<<dim3((B * S + 255) / 256), blk, 0, stream>>>(mask, outm, B * S);

  // 3) logits = Q x K^T per batch (f32 out)
  wmma_gemm<false, false, false><<<dim3(S / 128, S / 128, B), blk, 0, stream>>>(
      Qb, Kb, Lg, nullptr, nullptr, DM, DM, DM, S,
      (long long)S * DM, (long long)S * DM, (long long)S * S, 1.0f);

  // 4) masked softmax -> bf16 probabilities
  softmax_rows<<<dim3(S, B), blk, 0, stream>>>(Lg, mask, Pb, S);

  // 5) contexts = P x (Vt)^T per batch (bf16 out)
  wmma_gemm<false, true, false><<<dim3(DM / 128, S / 128, B), blk, 0, stream>>>(
      Pb, Vt, Ctx, nullptr, nullptr, S, S, S, DM,
      (long long)S * S, (long long)DM * S, (long long)S * DM, 1.0f);

  // 6) outputs = contexts x Wo (f32 out, duplicated to segments 2 and 3)
  wmma_gemm<false, false, false><<<dim3(DOUT / 128, (B * S) / 128, 1), blk, 0, stream>>>(
      Ctx, Wot, out0, out2, out3, DM, DM, DM, DOUT, 0, 0, 0, 1.0f);
}
